// VisionTransformer_40235253629236
// MI455X (gfx1250) — matrix operand order repaired and checked by reading, not compiled, not run on hardware
//
#include <hip/hip_runtime.h>
#include <hip/hip_bf16.h>

// ---------------- problem constants ----------------
#define BATCH 32
#define DIM   32
#define E     256
#define HEADS 8
#define LAYERS 4
#define FFD   1024
#define NP    4096
#define DH    32
#define SEQ   9          // 1 cls + 8 window tokens
#define NWIN  512

typedef __attribute__((ext_vector_type(16))) _Float16 v16h;
typedef __attribute__((ext_vector_type(8)))  float    v8f;

union AFrag { v16h v; float4 f4[2]; };

// weight sizes (elements) per layer, swizzled f16
#define QKV_SZ (768*256)
#define OUT_SZ (256*256)
#define FF1_SZ (1024*256)
#define FF2_SZ (1024*256)
#define LYR_SZ (QKV_SZ + OUT_SZ + FF1_SZ + FF2_SZ)

// workspace layout (bytes)
#define TOK_OFF  0ull
#define TOK_BYTES ((size_t)BATCH * NP * E * 2)
#define W_OFF    (TOK_OFF + TOK_BYTES)
#define W_BYTES  ((size_t)LAYERS * LYR_SZ * 2)
#define CLS_OFF  (W_OFF + W_BYTES)

// ---------------------------------------------------------------------------
// CDNA5 async global->LDS copy (ASYNCcnt-tracked, no VGPR data path).
// Each lane moves 16 contiguous bytes to its own LDS offset.
// ---------------------------------------------------------------------------
__device__ __forceinline__ void async_b128_to_lds(void* lds_ptr, const void* gptr)
{
    unsigned lds_off = (unsigned)(uintptr_t)lds_ptr;   // LDS aperture: low 32b = offset
    asm volatile("global_load_async_to_lds_b128 %0, %1, off"
                 :: "v"(lds_off), "v"(gptr) : "memory");
}
__device__ __forceinline__ void wait_async0()
{
    asm volatile("s_wait_asynccnt 0" ::: "memory");
}

// ---------------------------------------------------------------------------
// Token embedding: 2x2x2 stride-2 conv (circular pad) + pos_emb, store f16.
// ---------------------------------------------------------------------------
__global__ void __launch_bounds__(256) build_tokens(
    const float* __restrict__ x, const float* __restrict__ conv_w,
    const float* __restrict__ conv_b, const float* __restrict__ pos_emb,
    _Float16* __restrict__ tok)
{
    int bp = blockIdx.x;
    int b = bp >> 12;
    int p = bp & 4095;
    int z = p >> 8, yy = (p >> 4) & 15, xx = p & 15;
    __shared__ float xv[8];
    int tid = threadIdx.x;
    if (tid < 8) {
        int di = tid >> 2, dj = (tid >> 1) & 1, dk = tid & 1;
        int zi = (2 * z + di) & 31, yi = (2 * yy + dj) & 31, xi = (2 * xx + dk) & 31;
        xv[tid] = x[(size_t)b * (DIM * DIM * DIM) + (size_t)zi * (DIM * DIM) + yi * DIM + xi];
    }
    __syncthreads();
    float acc = conv_b[tid];
#pragma unroll
    for (int m = 0; m < 8; ++m) acc += xv[m] * conv_w[tid * 8 + m];
    acc += pos_emb[(size_t)(p + 1) * E + tid];
    tok[((size_t)b * NP + p) * E + tid] = (_Float16)acc;
}

__global__ void __launch_bounds__(256) init_cls(
    const float* __restrict__ cls_tok, const float* __restrict__ pos_emb,
    float* __restrict__ cls)
{
    cls[blockIdx.x * E + threadIdx.x] = cls_tok[threadIdx.x] + pos_emb[threadIdx.x];
}

// ---------------------------------------------------------------------------
// Swizzle f32 weight (N,K row-major, used as y @ W^T) into WMMA B-fragment
// order: tile (nt,kt) -> 512 f16; lane L owns 16 contiguous halves (32B).
// ---------------------------------------------------------------------------
__global__ void __launch_bounds__(256) swizzle_w(
    const float* __restrict__ src, _Float16* __restrict__ dst, int N, int K)
{
    int idx = blockIdx.x * 256 + threadIdx.x;
    if (idx >= N * K) return;
    int e    = idx & 15;
    int lane = (idx >> 4) & 31;
    int tile = idx >> 9;
    int KT   = K >> 5;
    int nt = tile / KT, kt = tile - nt * KT;
    int n = nt * 16 + (lane & 15);
    int k = kt * 32 + ((e >> 3) << 4) + ((lane >> 4) << 3) + (e & 7);
    dst[idx] = (_Float16)src[(size_t)n * K + k];
}

// ---------------------------------------------------------------------------
// One-row-tile WMMA GEMM, compile-time configured. D(.. x N) = A(16xK) B(KxN).
//   MODE 0: f32 accumulate into 9-row dst (residual), rows>=SEQ dropped
//   MODE 1: f16 store, full 16 rows (branchless)
//   MODE 2: f16 store + relu, full 16 rows (branchless)
// ---------------------------------------------------------------------------
#define WB_F32_ACC 0
#define WB_F16     1
#define WB_F16_RELU 2

template<int K, int N, int MODE, int DSTRIDE>
__device__ __forceinline__ void gemm_rowtile(
    const _Float16* __restrict__ Bw,   // swizzled weights (L2-resident)
    const _Float16* __restrict__ sAh,  // 16 x K staged activations (LDS)
    const float* __restrict__ bias,    // length N
    float* __restrict__ dstF, _Float16* __restrict__ dstH)
{
    constexpr int KT = K >> 5;
    int lane = threadIdx.x & 31;
    int wave = threadIdx.x >> 5;
    int row  = lane & 15;
    int kb   = (lane >> 4) << 3;           // A-frag K sub-offset: 0 or 8
    for (int nt = wave; nt < (N >> 4); nt += 8) {
        v8f acc = {};
        const _Float16* bp  = Bw + (size_t)nt * (KT * 512) + lane * 16;
        const _Float16* ap0 = sAh + row * K + kb;
#pragma unroll 8
        for (int kt = 0; kt < KT; ++kt) {
            AFrag a, bf;
            const _Float16* ap = ap0 + kt * 32;
            a.f4[0]  = *(const float4*)(ap);        // ds_load_b128 x2
            a.f4[1]  = *(const float4*)(ap + 16);
            bf.f4[0] = *(const float4*)(bp);        // global_load_b128 x2
            bf.f4[1] = *(const float4*)(bp + 8);
            bp += 512;
            acc = __builtin_amdgcn_wmma_f32_16x16x32_f16(
                false, a.v, false, bf.v, (short)0, acc, false, false);
        }
        int n  = nt * 16 + (lane & 15);
        float bv = bias[n];
        int r0 = (lane >> 4) << 3;                  // C/D rows r0..r0+7
#pragma unroll
        for (int r = 0; r < 8; ++r) {
            int m = r0 + r;
            float v = acc[r] + bv;
            if (MODE == WB_F32_ACC) {
                if (m < SEQ) dstF[m * DSTRIDE + n] += v;   // 9-row residual buffer
            } else if (MODE == WB_F16) {
                dstH[m * DSTRIDE + n] = (_Float16)v;       // branchless, 16 rows
            } else {
                dstH[m * DSTRIDE + n] = (_Float16)fmaxf(v, 0.0f);
            }
        }
    }
}

// stage f32 y (9 rows) -> 16xK f16 with zero padding rows
__device__ __forceinline__ void stage_f16(
    const float* __restrict__ src, _Float16* __restrict__ dst)
{
    for (int i = threadIdx.x; i < 16 * E; i += 256) {
        int m = i >> 8;
        dst[i] = (m < SEQ) ? (_Float16)src[i] : (_Float16)0.0f;
    }
}

// in-place LayerNorm over E=256 columns for rows 0..SEQ-1 (blockDim == 256)
__device__ __forceinline__ void layernorm_rows(
    float* __restrict__ y, const float* __restrict__ w,
    const float* __restrict__ b, float* __restrict__ red)
{
    int tid = threadIdx.x, lane = tid & 31, wv = tid >> 5;
    for (int m = 0; m < SEQ; ++m) {
        float v = y[m * E + tid];
        float s = v, s2 = v * v;
#pragma unroll
        for (int off = 16; off; off >>= 1) {
            s  += __shfl_down(s,  off, 32);
            s2 += __shfl_down(s2, off, 32);
        }
        if (lane == 0) { red[wv] = s; red[8 + wv] = s2; }
        __syncthreads();
        if (tid == 0) {
            float ts = 0.0f, ts2 = 0.0f;
#pragma unroll
            for (int i = 0; i < 8; ++i) { ts += red[i]; ts2 += red[8 + i]; }
            float mean = ts * (1.0f / E);
            float var  = ts2 * (1.0f / E) - mean * mean;
            red[16] = mean;
            red[17] = rsqrtf(var + 1e-5f);
        }
        __syncthreads();
        float mean = red[16], inv = red[17];
        y[m * E + tid] = (v - mean) * inv * w[tid] + b[tid];
        __syncthreads();
    }
}

// ---------------------------------------------------------------------------
// Sequential window scan: one block per batch chain, 256 threads = 8 waves.
// Window tokens double-buffered in LDS via async global->LDS DMA.
// ---------------------------------------------------------------------------
__global__ void __launch_bounds__(256) window_scan(
    const _Float16* __restrict__ tok, const _Float16* __restrict__ wts,
    const float* __restrict__ qkv_b, const float* __restrict__ out_b,
    const float* __restrict__ ff1_b, const float* __restrict__ ff2_b,
    const float* __restrict__ ln1_w, const float* __restrict__ ln1_b,
    const float* __restrict__ ln2_w, const float* __restrict__ ln2_b,
    float* __restrict__ cls)
{
    __shared__ float    sY[SEQ * E];                       // 9.2 KB residual stream
    __shared__ _Float16 sA[16 * E];                        // 8 KB A-operand staging
    __shared__ __align__(16) unsigned char sPool[16 * FFD * 2]; // 32 KB shared pool
    __shared__ __align__(16) _Float16 sTok[2][8 * E];      // 8 KB token double buffer

    _Float16* sQKVh = (_Float16*)sPool;                    // 16 x 768 f16 (q|k|v)
    float*    sSc   = (float*)(sPool + 16 * 768 * 2);      // 8 x 9 x 9 f32 scores
    _Float16* sH    = (_Float16*)sPool;                    // 16 x 1024 f16 (ff phase)
    float*    sRed  = (float*)(sPool + 16 * FFD * 2 - 128);// LN scratch (pool tail)

    int b = blockIdx.x;
    int tid = threadIdx.x;
    const _Float16* tb = tok + (size_t)b * NP * E;
    const float inv_sqrt_dh = 0.17677669529663687f;        // 1/sqrt(32)

    // thread -> (token d, 8-half chunk) mapping for async token copies
    int td  = tid >> 5;            // wave id = token 0..7
    int te8 = (tid & 31) * 8;      // halves within token

    // cls -> row 0
    sY[tid] = cls[b * E + tid];

    // kick off async copy of window 0 tokens into buffer 0
    {
        int p0 = (td >> 2) * 256 + ((td >> 1) & 1) * 16 + (td & 1);
        async_b128_to_lds(&sTok[0][td * E + te8], tb + (size_t)p0 * E + te8);
    }

    for (int w = 0; w < NWIN; ++w) {
        int cur = w & 1;
        wait_async0();                 // our wave's async writes are in LDS
        __syncthreads();               // everyone's are visible

        // consume: convert f16 tokens -> f32 rows 1..8 of sY
        for (int i = tid; i < 8 * E; i += 256)
            sY[E + i] = (float)sTok[cur][i];

        // produce: start async fetch of window w+1 into the other buffer
        if (w + 1 < NWIN) {
            int wn = w + 1;
            int wi = (wn >> 6) & 7, wj = (wn >> 3) & 7, wk = wn & 7;
            int p = (2 * wi + (td >> 2)) * 256 + (2 * wj + ((td >> 1) & 1)) * 16
                  + (2 * wk + (td & 1));
            async_b128_to_lds(&sTok[1 - cur][td * E + te8], tb + (size_t)p * E + te8);
        }
        __syncthreads();

        for (int l = 0; l < LAYERS; ++l) {
            const _Float16* wl = wts + (size_t)l * LYR_SZ;

            // ---------------- attention ----------------
            stage_f16(sY, sA);
            __syncthreads();
            gemm_rowtile<256, 768, WB_F16, 768>(wl, sA, qkv_b + l * 768, nullptr, sQKVh);
            __syncthreads();

            // scores = q k^T / sqrt(dh)   (8 heads x 9 x 9)
            for (int idx = tid; idx < HEADS * SEQ * SEQ; idx += 256) {
                int h = idx / (SEQ * SEQ);
                int r = idx - h * (SEQ * SEQ);
                int s = r / SEQ, t = r - s * SEQ;
                const _Float16* qp = sQKVh + s * 768 + h * DH;
                const _Float16* kp = sQKVh + t * 768 + 256 + h * DH;
                float d = 0.0f;
#pragma unroll
                for (int c = 0; c < DH; ++c) d += (float)qp[c] * (float)kp[c];
                sSc[idx] = d * inv_sqrt_dh;
            }
            __syncthreads();
            // softmax over last dim
            for (int idx = tid; idx < HEADS * SEQ; idx += 256) {
                float* row = sSc + idx * SEQ;
                float mx = row[0];
#pragma unroll
                for (int t = 1; t < SEQ; ++t) mx = fmaxf(mx, row[t]);
                float sum = 0.0f;
#pragma unroll
                for (int t = 0; t < SEQ; ++t) { float e2 = __expf(row[t] - mx); row[t] = e2; sum += e2; }
                float inv = 1.0f / sum;
#pragma unroll
                for (int t = 0; t < SEQ; ++t) row[t] *= inv;
            }
            __syncthreads();
            // o = att @ v  -> staged straight into f16 A operand (pad rows zero)
            for (int idx = tid; idx < 16 * E; idx += 256) {
                int m = idx >> 8, c = idx & 255;
                float v = 0.0f;
                if (m < SEQ) {
                    int h = c >> 5;
                    const float* att = sSc + (h * SEQ + m) * SEQ;
#pragma unroll
                    for (int t = 0; t < SEQ; ++t)
                        v += att[t] * (float)sQKVh[t * 768 + 512 + c];
                }
                sA[idx] = (_Float16)v;
            }
            __syncthreads();
            // out projection, accumulated onto residual
            gemm_rowtile<256, 256, WB_F32_ACC, 256>(wl + QKV_SZ, sA, out_b + l * E, sY, nullptr);
            __syncthreads();
            layernorm_rows(sY, ln1_w + l * E, ln1_b + l * E, sRed);

            // ---------------- feed-forward ----------------
            stage_f16(sY, sA);
            __syncthreads();
            gemm_rowtile<256, FFD, WB_F16_RELU, FFD>(wl + QKV_SZ + OUT_SZ, sA,
                                                     ff1_b + l * FFD, nullptr, sH);
            __syncthreads();
            gemm_rowtile<FFD, 256, WB_F32_ACC, 256>(wl + QKV_SZ + OUT_SZ + FF1_SZ, sH,
                                                    ff2_b + l * E, sY, nullptr);
            __syncthreads();
            layernorm_rows(sY, ln2_w + l * E, ln2_b + l * E, sRed);
        }
        // row 0 of sY is the updated cls; rows 1..8 refilled next window
    }
    cls[b * E + tid] = sY[tid];
}

// ---------------------------------------------------------------------------
// Final LayerNorm on cls + 10-class head.
// ---------------------------------------------------------------------------
__global__ void __launch_bounds__(256) head_kernel(
    const float* __restrict__ cls, const float* __restrict__ norm_w,
    const float* __restrict__ norm_b, const float* __restrict__ head_w,
    const float* __restrict__ head_b, float* __restrict__ out)
{
    __shared__ float sRed[32];
    __shared__ float sV[E];
    int b = blockIdx.x, tid = threadIdx.x;
    float v = cls[b * E + tid];
    float s = v, s2 = v * v;
#pragma unroll
    for (int off = 16; off; off >>= 1) {
        s  += __shfl_down(s,  off, 32);
        s2 += __shfl_down(s2, off, 32);
    }
    int lane = tid & 31, wv = tid >> 5;
    if (lane == 0) { sRed[wv] = s; sRed[8 + wv] = s2; }
    __syncthreads();
    if (tid == 0) {
        float ts = 0.0f, ts2 = 0.0f;
#pragma unroll
        for (int i = 0; i < 8; ++i) { ts += sRed[i]; ts2 += sRed[8 + i]; }
        float mean = ts * (1.0f / E);
        float var  = ts2 * (1.0f / E) - mean * mean;
        sRed[16] = mean;
        sRed[17] = rsqrtf(var + 1e-5f);
    }
    __syncthreads();
    sV[tid] = (v - sRed[16]) * sRed[17] * norm_w[tid] + norm_b[tid];
    __syncthreads();
    if (tid < 10) {
        float acc = head_b[tid];
        for (int e = 0; e < E; ++e) acc += sV[e] * head_w[tid * E + e];
        out[b * 10 + tid] = acc;
    }
}

// ---------------------------------------------------------------------------
extern "C" void kernel_launch(void* const* d_in, const int* in_sizes, int n_in,
                              void* d_out, int out_size, void* d_ws, size_t ws_size,
                              hipStream_t stream)
{
    const float* x        = (const float*)d_in[0];
    const float* conv_w   = (const float*)d_in[1];
    const float* conv_b   = (const float*)d_in[2];
    const float* cls_tok  = (const float*)d_in[3];
    const float* pos_emb  = (const float*)d_in[4];
    const float* ln1_w    = (const float*)d_in[5];
    const float* ln1_b    = (const float*)d_in[6];
    const float* ln2_w    = (const float*)d_in[7];
    const float* ln2_b    = (const float*)d_in[8];
    const float* qkv_w    = (const float*)d_in[9];
    const float* qkv_b    = (const float*)d_in[10];
    const float* out_w    = (const float*)d_in[11];
    const float* out_b    = (const float*)d_in[12];
    const float* ff1_w    = (const float*)d_in[13];
    const float* ff1_b    = (const float*)d_in[14];
    const float* ff2_w    = (const float*)d_in[15];
    const float* ff2_b    = (const float*)d_in[16];
    const float* norm_w   = (const float*)d_in[17];
    const float* norm_b   = (const float*)d_in[18];
    const float* head_w   = (const float*)d_in[19];
    const float* head_b   = (const float*)d_in[20];

    unsigned char* ws = (unsigned char*)d_ws;
    _Float16* tok = (_Float16*)(ws + TOK_OFF);
    _Float16* wts = (_Float16*)(ws + W_OFF);
    float*    cls = (float*)(ws + CLS_OFF);

    // 1) token embedding + positional embedding (f16)
    build_tokens<<<BATCH * NP, 256, 0, stream>>>(x, conv_w, conv_b, pos_emb, tok);
    init_cls<<<BATCH, 256, 0, stream>>>(cls_tok, pos_emb, cls);

    // 2) swizzle weights into WMMA B-fragment layout, f16 (L2-resident)
    for (int l = 0; l < LAYERS; ++l) {
        _Float16* wl = wts + (size_t)l * LYR_SZ;
        swizzle_w<<<(QKV_SZ + 255) / 256, 256, 0, stream>>>(
            qkv_w + (size_t)l * QKV_SZ, wl, 768, 256);
        swizzle_w<<<(OUT_SZ + 255) / 256, 256, 0, stream>>>(
            out_w + (size_t)l * OUT_SZ, wl + QKV_SZ, 256, 256);
        swizzle_w<<<(FF1_SZ + 255) / 256, 256, 0, stream>>>(
            ff1_w + (size_t)l * FF1_SZ, wl + QKV_SZ + OUT_SZ, 1024, 256);
        swizzle_w<<<(FF2_SZ + 255) / 256, 256, 0, stream>>>(
            ff2_w + (size_t)l * FF2_SZ, wl + QKV_SZ + OUT_SZ + FF1_SZ, 256, 1024);
    }

    // 3) the sequential 512-window scan: 32 independent batch chains
    window_scan<<<BATCH, 256, 0, stream>>>(
        tok, wts, qkv_b, out_b, ff1_b, ff2_b,
        ln1_w, ln1_b, ln2_w, ln2_b, cls);

    // 4) final LN + classification head
    head_kernel<<<BATCH, 256, 0, stream>>>(
        cls, norm_w, norm_b, head_w, head_b, (float*)d_out);
}
